// HeteroIncidentGATv2_64673617543675
// MI455X (gfx1250) — matrix operand order repaired and checked by reading, not compiled, not run on hardware
//
#include <hip/hip_runtime.h>
#include <hip/hip_bf16.h>
#include <stdint.h>

typedef __attribute__((ext_vector_type(16))) _Float16 v16h;
typedef __attribute__((ext_vector_type(8)))  float    v8f;

#define HCN 128   // H*C
#define DIN 64

// ---------------- order-preserving float <-> uint encoding for atomic max --
__device__ __forceinline__ unsigned enc_f32(float f) {
  unsigned u = __float_as_uint(f);
  return (u & 0x80000000u) ? ~u : (u | 0x80000000u);
}
__device__ __forceinline__ float dec_f32(unsigned u) {
  unsigned b = (u & 0x80000000u) ? (u & 0x7fffffffu) : ~u;
  return __uint_as_float(b);
}

// ---------------- pack 16 [64x128] f32 weight matrices into f16 B-fragment
// layout: wp[((mat*8+t)*2+kbi)*512 + lane*16 + j] = W[kbi*32 + (lane>=16?16:0)+j][t*16 + lane%16]
__global__ void pack_w(const float* __restrict__ Wl, const float* __restrict__ Wr,
                       const float* __restrict__ Wsp, _Float16* __restrict__ wp) {
  int idx  = blockIdx.x * 256 + threadIdx.x;          // 0 .. 131071
  int j    = idx & 15;
  int lane = (idx >> 4) & 31;
  int kbi  = (idx >> 9) & 1;
  int t    = (idx >> 10) & 7;
  int mat  = idx >> 13;
  const float* W = (mat < 6)  ? (Wl  + (size_t)mat        * DIN * HCN)
                 : (mat < 12) ? (Wr  + (size_t)(mat - 6)  * DIN * HCN)
                              : (Wsp + (size_t)(mat - 12) * DIN * HCN);
  int k = kbi * 32 + ((lane >> 4) << 4) + j;
  int n = t * 16 + (lane & 15);
  wp[idx] = (_Float16)W[k * HCN + n];
}

// ---------------- WMMA GEMM: Y[N,128] = f16(X[N,64]) @ f16(W[64,128]) + bias
// one wave per 16-row tile; 8 col-tiles x 2 k-steps = 16 WMMAs.
// Wave-uniform full-tile fast path -> branchless, unguarded stores.
__global__ void gemm_proj(const float* __restrict__ X, const _Float16* __restrict__ Wp,
                          const float* __restrict__ bias, float* __restrict__ Y, int N) {
  int wave = threadIdx.x >> 5, lane = threadIdx.x & 31;
  int m0 = (blockIdx.x * 8 + wave) * 16;
  if (m0 >= N) return;                                 // wave-uniform exit, EXEC stays full
  bool full = (m0 + 16 <= N);                          // wave-uniform
  int row = m0 + (lane & 15);
  if (!full && row >= N) row = N - 1;                  // clamp only on tail tile
  const float* xp = X + (size_t)row * DIN;
  int base = (lane >> 4) << 4;                         // A layout: hi half-wave holds K+16..31
  v16h a0, a1;
#pragma unroll
  for (int j = 0; j < 16; ++j) {
    int k = base + j;
    a0[j] = (_Float16)xp[k];
    a1[j] = (_Float16)xp[32 + k];
  }
  int col0   = lane & 15;
  int rowoff = (lane >> 4) << 3;
#pragma unroll
  for (int t = 0; t < 8; ++t) {
    v16h b0 = ((const v16h*)(Wp + t * 1024))[lane];
    v16h b1 = ((const v16h*)(Wp + t * 1024 + 512))[lane];
    v8f c = {0.f, 0.f, 0.f, 0.f, 0.f, 0.f, 0.f, 0.f};
    c = __builtin_amdgcn_wmma_f32_16x16x32_f16(false, a0, false, b0, (short)0, c, false, false);
    c = __builtin_amdgcn_wmma_f32_16x16x32_f16(false, a1, false, b1, (short)0, c, false, false);
    int cc = t * 16 + col0;
    float bv = bias[cc];
    float* yp = Y + (size_t)(m0 + rowoff) * HCN + cc;
    if (full) {
#pragma unroll
      for (int r2 = 0; r2 < 8; ++r2)
        yp[(size_t)r2 * HCN] = c[r2] + bv;
    } else {
#pragma unroll
      for (int r2 = 0; r2 < 8; ++r2) {
        int orow = m0 + rowoff + r2;
        if (orow < N) Y[(size_t)orow * HCN + cc] = c[r2] + bv;
      }
    }
  }
}

// ---------------- small fills
__global__ void zero_u32(unsigned* __restrict__ p, int n) {
  int i = blockIdx.x * 256 + threadIdx.x;
  if (i < n) p[i] = 0u;
}
__global__ void init_agg(float* __restrict__ agg, const float* __restrict__ b1,
                         const float* __restrict__ b2, int n128) {
  int i = blockIdx.x * 256 + threadIdx.x;
  if (i >= n128) return;
  int c = i & (HCN - 1);
  float v = b1[c];
  if (b2) v += b2[c];
  agg[i] = v;
}

// ---------------- pass A: e = <leaky(xl[s]+xr[d]), att>, atomic segment max
__global__ void edge_scores(const float* __restrict__ xl, const float* __restrict__ xr,
                            const int* __restrict__ si, const int* __restrict__ di,
                            const float* __restrict__ att, float* __restrict__ ebuf,
                            unsigned* __restrict__ mbuf, int E) {
  int wave = threadIdx.x >> 5, lane = threadIdx.x & 31;
  int e = blockIdx.x * 8 + wave;
  if (e >= E) return;
  int s = si[e], d = di[e];
  const float* pl = xl + (size_t)s * HCN;
  const float* pr = xr + (size_t)d * HCN;
  float acc = 0.f;
#pragma unroll
  for (int k = 0; k < 4; ++k) {
    int hc = lane * 4 + k;
    float f = pl[hc] + pr[hc];
    f = f > 0.f ? f : 0.2f * f;
    acc += f * att[hc];
  }
  acc += __shfl_xor(acc, 1, 32);
  acc += __shfl_xor(acc, 2, 32);
  acc += __shfl_xor(acc, 4, 32);
  if ((lane & 7) == 0) {
    int h = lane >> 3;
    ebuf[(size_t)e * 4 + h] = acc;
    atomicMax(mbuf + (size_t)d * 4 + h, enc_f32(acc));
  }
}

// ---------------- pass B: ee = exp(e - m[d]) (m non-finite -> 0), atomic denom
__global__ void edge_exp(const int* __restrict__ di, float* __restrict__ ebuf,
                         const unsigned* __restrict__ mbuf, float* __restrict__ den, int E4) {
  int i = blockIdx.x * 256 + threadIdx.x;
  if (i >= E4) return;
  int e = i >> 2, h = i & 3;
  int d = di[e];
  float mm = dec_f32(mbuf[(size_t)d * 4 + h]);
  unsigned mb = __float_as_uint(mm);
  if ((mb & 0x7f800000u) == 0x7f800000u) mm = 0.f;    // inf/nan -> 0 (matches reference)
  float ee = __expf(ebuf[i] - mm);
  ebuf[i] = ee;
  atomicAdd(den + (size_t)d * 4 + h, ee);
}

// ---------------- pass C: alpha -> d_out, atomic add alpha * xl[s] into agg[d]
__global__ void edge_agg(const float* __restrict__ xl, const int* __restrict__ si,
                         const int* __restrict__ di, const float* __restrict__ ebuf,
                         const float* __restrict__ den, float* __restrict__ alpha_out,
                         float* __restrict__ agg, int E) {
  int wave = threadIdx.x >> 5, lane = threadIdx.x & 31;
  int e = blockIdx.x * 8 + wave;
  if (e >= E) return;
  int s = si[e], d = di[e];
  int h = lane >> 3;
  float a = ebuf[(size_t)e * 4 + h] / (den[(size_t)d * 4 + h] + 1e-16f);
  if ((lane & 7) == 0) alpha_out[(size_t)e * 4 + h] = a;
  const float* pl = xl + (size_t)s * HCN;
  float* pa = agg + (size_t)d * HCN;
#pragma unroll
  for (int k = 0; k < 4; ++k) {
    int hc = lane * 4 + k;
    atomicAdd(pa + hc, a * pl[hc]);
  }
}

// ---------------- head: sigmoid(leaky(self+agg, .01) . wout + bout)
__global__ void head_k(const float* __restrict__ selfb, const float* __restrict__ agg,
                       const float* __restrict__ wout, const float* __restrict__ bout,
                       float* __restrict__ out, int N) {
  int wave = threadIdx.x >> 5, lane = threadIdx.x & 31;
  int n = blockIdx.x * 8 + wave;
  if (n >= N) return;
  float acc = 0.f;
#pragma unroll
  for (int k = 0; k < 4; ++k) {
    int hc = lane * 4 + k;
    float v = selfb[(size_t)n * HCN + hc] + agg[(size_t)n * HCN + hc];
    v = v > 0.f ? v : 0.01f * v;
    acc += v * wout[hc];
  }
  for (int o = 1; o < 32; o <<= 1) acc += __shfl_xor(acc, o, 32);
  if (lane == 0) out[n] = 1.f / (1.f + __expf(-(acc + bout[0])));
}

extern "C" void kernel_launch(void* const* d_in, const int* in_sizes, int n_in,
                              void* d_out, int out_size, void* d_ws, size_t ws_size,
                              hipStream_t stream) {
  (void)in_sizes; (void)n_in; (void)out_size; (void)ws_size;
  const float* x_host = (const float*)d_in[0];
  const float* x_vm   = (const float*)d_in[1];
  const float* x_job  = (const float*)d_in[2];
  const float* x_sw   = (const float*)d_in[3];
  const float* Wl   = (const float*)d_in[4];
  const float* bl   = (const float*)d_in[5];
  const float* Wr   = (const float*)d_in[6];
  const float* br   = (const float*)d_in[7];
  const float* attw = (const float*)d_in[8];
  const float* bc   = (const float*)d_in[9];
  const float* Wsp  = (const float*)d_in[10];
  const float* bsp  = (const float*)d_in[11];
  const float* Wout = (const float*)d_in[12];
  const float* bout = (const float*)d_in[13];
  const int* src_hs = (const int*)d_in[14];
  const int* dst_hs = (const int*)d_in[15];
  const int* src_vh = (const int*)d_in[16];
  const int* dst_vh = (const int*)d_in[17];
  const int* src_jv = (const int*)d_in[18];
  const int* dst_jv = (const int*)d_in[19];
  float* out = (float*)d_out;

  const int N_HOST = 100000, N_VM = 300000, N_JOB = 500000, N_SW = 2000;
  const int E_HS = 200000, E_VH = 300000, E_JV = 500000;
  const int NMAX = 500000;

  // ---- workspace layout (bytes); relations processed sequentially reuse xl/xr.
  // NOTE: mbuf and den are adjacent so they can be zeroed with one launch.
  char* ws = (char*)d_ws;
  size_t off = 0;
  _Float16* wp = (_Float16*)(ws + off); off += (size_t)16 * 8192 * 2;     // 256 KB
  float* xl_buf = (float*)(ws + off);  off += (size_t)NMAX * HCN * 4;     // 256 MB
  float* xr_buf = (float*)(ws + off);  off += (size_t)NMAX * HCN * 4;     // 256 MB (reused as self)
  float* ebuf   = (float*)(ws + off);  off += (size_t)NMAX * 4 * 4;       // 8 MB
  unsigned* mbuf= (unsigned*)(ws + off); off += (size_t)NMAX * 4 * 4;     // 8 MB
  float* den    = (float*)(ws + off);  off += (size_t)NMAX * 4 * 4;       // 8 MB
  float* agg_host = (float*)(ws + off); off += (size_t)N_HOST * HCN * 4;
  float* agg_vm   = (float*)(ws + off); off += (size_t)N_VM   * HCN * 4;
  float* agg_job  = (float*)(ws + off); off += (size_t)N_JOB  * HCN * 4;
  float* agg_sw   = (float*)(ws + off); off += (size_t)N_SW   * HCN * 4;
  float* self_buf = xr_buf;

  // ---- pack weights to f16 fragment layout (16 * 8192 halves)
  pack_w<<<512, 256, 0, stream>>>(Wl, Wr, Wsp, wp);

  // ---- seed agg buffers with summed conv biases
  init_agg<<<(N_HOST*HCN + 255)/256, 256, 0, stream>>>(agg_host, bc + 3*HCN, bc + 1*HCN, N_HOST*HCN);
  init_agg<<<(N_VM  *HCN + 255)/256, 256, 0, stream>>>(agg_vm,   bc + 4*HCN, bc + 2*HCN, N_VM*HCN);
  init_agg<<<(N_JOB *HCN + 255)/256, 256, 0, stream>>>(agg_job,  bc + 5*HCN, nullptr,    N_JOB*HCN);
  init_agg<<<(N_SW  *HCN + 255)/256, 256, 0, stream>>>(agg_sw,   bc + 0*HCN, nullptr,    N_SW*HCN);

  struct Rel {
    const float* xs; int Ns; const float* xd; int Nd;
    const int* si; const int* di; int E; float* agg; size_t aoff;
  };
  const Rel rels[6] = {
    { x_host, N_HOST, x_sw,   N_SW,   src_hs, dst_hs, E_HS, agg_sw,   902000u },  // r0 host->switch
    { x_vm,   N_VM,   x_host, N_HOST, src_vh, dst_vh, E_VH, agg_host, 1702000u }, // r1 vm->host
    { x_job,  N_JOB,  x_vm,   N_VM,   src_jv, dst_jv, E_JV, agg_vm,   2902000u }, // r2 job->vm
    { x_sw,   N_SW,   x_host, N_HOST, dst_hs, src_hs, E_HS, agg_host, 4902000u }, // r3 rev switch->host
    { x_host, N_HOST, x_vm,   N_VM,   dst_vh, src_vh, E_VH, agg_vm,   5702000u }, // r4 rev host->vm
    { x_vm,   N_VM,   x_job,  N_JOB,  dst_jv, src_jv, E_JV, agg_job,  6902000u }, // r5 rev vm->job
  };

  for (int r = 0; r < 6; ++r) {
    const Rel& R = rels[r];
    gemm_proj<<<(R.Ns + 127)/128, 256, 0, stream>>>(R.xs, wp + (size_t)r * 8192,
                                                    bl + r*HCN, xl_buf, R.Ns);
    gemm_proj<<<(R.Nd + 127)/128, 256, 0, stream>>>(R.xd, wp + (size_t)(6 + r) * 8192,
                                                    br + r*HCN, xr_buf, R.Nd);
    zero_u32<<<(R.Nd*8 + 255)/256, 256, 0, stream>>>(mbuf, R.Nd*8);   // first Nd*8 words from mbuf; covers all of den only if Nd == NMAX
    edge_scores<<<(R.E + 7)/8, 256, 0, stream>>>(xl_buf, xr_buf, R.si, R.di,
                                                 attw + r*HCN, ebuf, mbuf, R.E);
    edge_exp<<<(R.E*4 + 255)/256, 256, 0, stream>>>(R.di, ebuf, mbuf, den, R.E*4);
    edge_agg<<<(R.E + 7)/8, 256, 0, stream>>>(xl_buf, R.si, R.di, ebuf, den,
                                              out + R.aoff, R.agg, R.E);
  }

  // ---- self projection + heads (node-type order: host, vm, job, switch)
  const float* xt[4]   = { x_host, x_vm, x_job, x_sw };
  const int    Nt[4]   = { N_HOST, N_VM, N_JOB, N_SW };
  float*       aggt[4] = { agg_host, agg_vm, agg_job, agg_sw };
  const size_t soff[4] = { 0u, 100000u, 400000u, 900000u };
  for (int t = 0; t < 4; ++t) {
    gemm_proj<<<(Nt[t] + 127)/128, 256, 0, stream>>>(xt[t], wp + (size_t)(12 + t) * 8192,
                                                     bsp + t*HCN, self_buf, Nt[t]);
    head_k<<<(Nt[t] + 7)/8, 256, 0, stream>>>(self_buf, aggt[t], Wout + t*HCN,
                                              bout + t, out + soff[t], Nt[t]);
  }
}
